// CrAKNVectorAttention3D_69904887710287
// MI455X (gfx1250) — compile-verified, matrix operand order checked by reading, not run on hardware
//
#include <hip/hip_runtime.h>
#include <hip/hip_bf16.h>
#include <math.h>

typedef __attribute__((ext_vector_type(4)))  float  v4f;
typedef __attribute__((ext_vector_type(8)))  float  v8f;
typedef __attribute__((ext_vector_type(8)))  __bf16 v8bf;
typedef __attribute__((ext_vector_type(16))) __bf16 v16bf;

#define C_DIM 256
#define N_DIM 512
#define LN_EPS 1e-5f

__device__ __forceinline__ float mishf(float x) {
    float sp = (x > 20.f) ? x : log1pf(__expf(x));
    return x * tanhf(sp);
}

// ---------------------------------------------------------------------------
// Kernel 1: q = mish(LN(feat@Wq+bq)), k = mish(LN(feat@Wk+bk)), v = feat@Wv+bv,
//           dist[n] = (sum|feat[n,:]| > 0)
// One block (256 threads) per row n; feat row cached in LDS; thread c computes
// output channel c for all three projections.
// ---------------------------------------------------------------------------
__global__ __launch_bounds__(256) void qkv_kernel(
    const float* __restrict__ feat,
    const float* __restrict__ Wq, const float* __restrict__ bq,
    const float* __restrict__ gq, const float* __restrict__ bbq,
    const float* __restrict__ Wk, const float* __restrict__ bk,
    const float* __restrict__ gk, const float* __restrict__ bbk,
    const float* __restrict__ Wv, const float* __restrict__ bv,
    float* __restrict__ q, float* __restrict__ kk, float* __restrict__ v,
    float* __restrict__ dist)
{
    __shared__ float fr[C_DIM];
    __shared__ float red1[C_DIM];
    __shared__ float red2[C_DIM];

    const int n = blockIdx.x;
    const int tid = threadIdx.x;

    float f = feat[n * C_DIM + tid];
    fr[tid] = f;
    red1[tid] = fabsf(f);
    __syncthreads();
    for (int s = 128; s > 0; s >>= 1) {
        if (tid < s) red1[tid] += red1[tid + s];
        __syncthreads();
    }
    float dn = (red1[0] > 0.f) ? 1.f : 0.f;
    if (tid == 0) dist[n] = dn;
    __syncthreads();

    float accq = bq[tid], acck = bk[tid], accv = bv[tid];
    for (int kx = 0; kx < C_DIM; ++kx) {
        float fv = fr[kx];
        accq = fmaf(fv, Wq[kx * C_DIM + tid], accq);
        acck = fmaf(fv, Wk[kx * C_DIM + tid], acck);
        accv = fmaf(fv, Wv[kx * C_DIM + tid], accv);
    }
    v[n * C_DIM + tid] = accv;

    // LN + mish (q)
    red1[tid] = accq; red2[tid] = accq * accq;
    __syncthreads();
    for (int s = 128; s > 0; s >>= 1) {
        if (tid < s) { red1[tid] += red1[tid + s]; red2[tid] += red2[tid + s]; }
        __syncthreads();
    }
    {
        float mu  = red1[0] * (1.f / C_DIM);
        float var = red2[0] * (1.f / C_DIM) - mu * mu;
        float y = (accq - mu) * rsqrtf(var + LN_EPS) * gq[tid] + bbq[tid];
        q[n * C_DIM + tid] = mishf(y);
    }
    __syncthreads();

    // LN + mish (k)
    red1[tid] = acck; red2[tid] = acck * acck;
    __syncthreads();
    for (int s = 128; s > 0; s >>= 1) {
        if (tid < s) { red1[tid] += red1[tid + s]; red2[tid] += red2[tid + s]; }
        __syncthreads();
    }
    {
        float mu  = red1[0] * (1.f / C_DIM);
        float var = red2[0] * (1.f / C_DIM) - mu * mu;
        float y = (acck - mu) * rsqrtf(var + LN_EPS) * gk[tid] + bbk[tid];
        kk[n * C_DIM + tid] = mishf(y);
    }
}

// ---------------------------------------------------------------------------
// Kernel 2: pre-swizzle W1 / W2 into fragment-ready bf16 B-operand layout.
// Frag addr = ((ntile*8 + kstep)*32 + lane)*16 + e ; element = W[k][n] with
//   n = ntile*16 + (lane&15)
//   k = kstep*32 + (e&7) + ((e>>3)<<4) + (lane>>4)*8
// so each lane's 16 B values are one contiguous 32-byte run.
// ---------------------------------------------------------------------------
__global__ __launch_bounds__(256) void wswz_kernel(
    const float* __restrict__ W1, const float* __restrict__ W2,
    __bf16* __restrict__ w1f, __bf16* __restrict__ w2f)
{
    int id   = blockIdx.x * 256 + threadIdx.x;   // 0 .. 131071
    int wsel = id >> 16;
    int rem  = id & 65535;
    int e    = rem & 15;
    int lane = (rem >> 4) & 31;
    int ks   = (rem >> 9) & 7;
    int nt   = (rem >> 12) & 15;
    int kidx = ks * 32 + (e & 7) + ((e >> 3) << 4) + (lane >> 4) * 8;
    int nidx = nt * 16 + (lane & 15);
    const float* W = wsel ? W2 : W1;
    __bf16* Dst    = wsel ? w2f : w1f;
    Dst[rem] = (__bf16)W[kidx * C_DIM + nidx];
}

// ---------------------------------------------------------------------------
// Per-wave 16x256 tile: A(16x256,bf16 in LDS) x B(256x256, frag-ready, held in
// VGPRs across the j-loop by the compiler). Each of 8 waves owns 2 N-tiles;
// 8 K-steps of v_wmma_f32_16x16x32_bf16 per tile.
// ---------------------------------------------------------------------------
__device__ __forceinline__ void gemm16(
    const __bf16* __restrict__ Alds,
    const __bf16* __restrict__ Bfrag,
    const float*  __restrict__ bias,
    float* __restrict__ Out, int tid)
{
    const int lane = tid & 31;
    const int w    = tid >> 5;
    const int m    = lane & 15;
    const int half = lane >> 4;
    const int nt0  = w * 2, nt1 = w * 2 + 1;

    v8f acc0 = {};
    v8f acc1 = {};
#pragma unroll
    for (int ks = 0; ks < 8; ++ks) {
        const __bf16* ap = Alds + m * C_DIM + ks * 32 + half * 8;
        v8bf alo = *(const v8bf*)ap;
        v8bf ahi = *(const v8bf*)(ap + 16);
        v16bf a;
#pragma unroll
        for (int t = 0; t < 8; ++t) { a[t] = alo[t]; a[t + 8] = ahi[t]; }

        const __bf16* bp0 = Bfrag + ((nt0 * 8 + ks) * 32 + lane) * 16;
        const __bf16* bp1 = Bfrag + ((nt1 * 8 + ks) * 32 + lane) * 16;
        v8bf b0lo = *(const v8bf*)bp0;
        v8bf b0hi = *(const v8bf*)(bp0 + 8);
        v8bf b1lo = *(const v8bf*)bp1;
        v8bf b1hi = *(const v8bf*)(bp1 + 8);
        v16bf b0, b1v;
#pragma unroll
        for (int t = 0; t < 8; ++t) {
            b0[t] = b0lo[t]; b0[t + 8] = b0hi[t];
            b1v[t] = b1lo[t]; b1v[t + 8] = b1hi[t];
        }
        acc0 = __builtin_amdgcn_wmma_f32_16x16x32_bf16(
            false, a, false, b0, (short)0, acc0, false, false);
        acc1 = __builtin_amdgcn_wmma_f32_16x16x32_bf16(
            false, a, false, b1v, (short)0, acc1, false, false);
    }
    const int n0 = nt0 * 16 + (lane & 15);
    const int n1 = nt1 * 16 + (lane & 15);
#pragma unroll
    for (int vr = 0; vr < 8; ++vr) {
        int M = vr + 8 * half;   // C/D layout: VGPR r -> M=r (lanes 0-15), M=r+8 (16-31)
        Out[M * C_DIM + n0] = acc0[vr] + bias[n0];
        Out[M * C_DIM + n1] = acc1[vr] + bias[n1];
    }
}

// ---------------------------------------------------------------------------
// Kernel 3: fused pairwise-MLP attention. One block per query i.
// Streams 32 j-tiles of 16: rel -> GEMM(W1) -> LN+mish -> GEMM(W2) ->
// per-channel online softmax with dist gating -> out[i,c] = acc/l.
// Elementwise stages use (row, 16-consecutive-channel) thread mapping so all
// LDS traffic is b128 (2x ds_store_b128 per thread instead of 16x b16).
// ---------------------------------------------------------------------------
__global__ __launch_bounds__(256) void cra_attn_kernel(
    const float* __restrict__ q, const float* __restrict__ k,
    const float* __restrict__ v, const float* __restrict__ dist,
    const __bf16* __restrict__ w1f, const __bf16* __restrict__ w2f,
    const float* __restrict__ b1, const float* __restrict__ ln_w_g,
    const float* __restrict__ ln_w_b, const float* __restrict__ b2,
    float* __restrict__ out)
{
    __shared__ __align__(16) float  qi[C_DIM];
    __shared__ __align__(16) __bf16 relA[16 * C_DIM];
    __shared__ __align__(16) __bf16 sA[16 * C_DIM];
    __shared__ __align__(16) float  tbuf[16 * C_DIM];
    __shared__ float redS[16][17];
    __shared__ float redS2[16][17];
    __shared__ float muS[16], rsS[16];
    __shared__ float mb[C_DIM], lb[C_DIM], ab[C_DIM];
    __shared__ float distsh[16];
    __shared__ __align__(16) float b1s[C_DIM];
    __shared__ __align__(16) float b2s[C_DIM];
    __shared__ __align__(16) float gs[C_DIM];
    __shared__ __align__(16) float bs[C_DIM];

    const int i   = blockIdx.x;
    const int tid = threadIdx.x;
    const int r   = tid >> 4;          // row 0..15
    const int c0  = (tid & 15) * 16;   // 16-channel segment base

    qi[tid]  = q[i * C_DIM + tid];
    b1s[tid] = b1[tid];
    b2s[tid] = b2[tid];
    gs[tid]  = ln_w_g[tid];
    bs[tid]  = ln_w_b[tid];
    mb[tid]  = -3.0e38f;
    lb[tid]  = 0.f;
    ab[tid]  = 0.f;
    __syncthreads();

    for (int jt = 0; jt < N_DIM / 16; ++jt) {
        const int j0 = jt * 16;
        if (tid < 16) distsh[tid] = dist[j0 + tid];

        // rel tile: relA[r][c] = bf16(k[j0+r,c] - q[i,c]); b128 in, b128 out
        {
            const v4f* kp = (const v4f*)(k + (j0 + r) * C_DIM + c0);
            const v4f* qp = (const v4f*)(qi + c0);
            v8bf p0, p1;
#pragma unroll
            for (int vv = 0; vv < 4; ++vv) {
                v4f kf = kp[vv];
                v4f qf = qp[vv];
#pragma unroll
                for (int e = 0; e < 4; ++e) {
                    __bf16 b = (__bf16)(kf[e] - qf[e]);
                    if (vv < 2) p0[vv * 4 + e] = b;
                    else        p1[(vv - 2) * 4 + e] = b;
                }
            }
            *(v8bf*)(relA + r * C_DIM + c0)     = p0;
            *(v8bf*)(relA + r * C_DIM + c0 + 8) = p1;
        }
        __syncthreads();

        gemm16(relA, w1f, b1s, tbuf, tid);     // t = rel @ W1 + b1
        __syncthreads();

        // LayerNorm stats per row (16 segments of 16 channels each)
        {
            const v4f* tp = (const v4f*)(tbuf + r * C_DIM + c0);
            float s1 = 0.f, s2 = 0.f;
#pragma unroll
            for (int vv = 0; vv < 4; ++vv) {
                v4f x = tp[vv];
#pragma unroll
                for (int e = 0; e < 4; ++e) { s1 += x[e]; s2 += x[e] * x[e]; }
            }
            redS[r][tid & 15] = s1; redS2[r][tid & 15] = s2;
        }
        __syncthreads();
        if (tid < 16) {
            float s1 = 0.f, s2 = 0.f;
            for (int seg = 0; seg < 16; ++seg) { s1 += redS[tid][seg]; s2 += redS2[tid][seg]; }
            float mu  = s1 * (1.f / C_DIM);
            float var = s2 * (1.f / C_DIM) - mu * mu;
            muS[tid] = mu;
            rsS[tid] = rsqrtf(var + LN_EPS);
        }
        __syncthreads();

        // LN + mish apply -> sA (bf16, b128 stores)
        {
            const float mu = muS[r], rs = rsS[r];
            const v4f* tp = (const v4f*)(tbuf + r * C_DIM + c0);
            const v4f* gp = (const v4f*)(gs + c0);
            const v4f* bp = (const v4f*)(bs + c0);
            v8bf p0, p1;
#pragma unroll
            for (int vv = 0; vv < 4; ++vv) {
                v4f x = tp[vv];
                v4f g = gp[vv];
                v4f b = bp[vv];
#pragma unroll
                for (int e = 0; e < 4; ++e) {
                    float y = (x[e] - mu) * rs * g[e] + b[e];
                    __bf16 bb = (__bf16)mishf(y);
                    if (vv < 2) p0[vv * 4 + e] = bb;
                    else        p1[(vv - 2) * 4 + e] = bb;
                }
            }
            *(v8bf*)(sA + r * C_DIM + c0)     = p0;
            *(v8bf*)(sA + r * C_DIM + c0 + 8) = p1;
        }
        __syncthreads();

        gemm16(sA, w2f, b2s, tbuf, tid);       // w = s @ W2 + b2
        __syncthreads();

        // per-channel online softmax over j, gated by dist[j]
        {
            const int c = tid;
            float mo = mb[c];
            float mn = mo;
            float wv[16], dv[16];
#pragma unroll
            for (int jj = 0; jj < 16; ++jj) {
                wv[jj] = tbuf[jj * C_DIM + c];
                dv[jj] = distsh[jj];
                if (dv[jj] > 0.f) mn = fmaxf(mn, wv[jj]);
            }
            float l = lb[c], a = ab[c];
            if (mn > mo) { float sc = __expf(mo - mn); l *= sc; a *= sc; }
#pragma unroll
            for (int jj = 0; jj < 16; ++jj) {
                if (dv[jj] > 0.f) {
                    float e = __expf(wv[jj] - mn);
                    l += e;
                    a += e * v[(j0 + jj) * C_DIM + c];
                }
            }
            mb[c] = mn; lb[c] = l; ab[c] = a;
        }
        __syncthreads();
    }

    float o = (dist[i] > 0.f) ? (ab[tid] / lb[tid]) : 0.f;
    out[i * C_DIM + tid] = o;
}

// ---------------------------------------------------------------------------
extern "C" void kernel_launch(void* const* d_in, const int* in_sizes, int n_in,
                              void* d_out, int out_size, void* d_ws, size_t ws_size,
                              hipStream_t stream)
{
    (void)in_sizes; (void)n_in; (void)out_size; (void)ws_size;
    const float* feat   = (const float*)d_in[0];
    const float* Wq     = (const float*)d_in[1];
    const float* bq     = (const float*)d_in[2];
    const float* ln_q_g = (const float*)d_in[3];
    const float* ln_q_b = (const float*)d_in[4];
    const float* Wk     = (const float*)d_in[5];
    const float* bk     = (const float*)d_in[6];
    const float* ln_k_g = (const float*)d_in[7];
    const float* ln_k_b = (const float*)d_in[8];
    const float* Wv     = (const float*)d_in[9];
    const float* bv     = (const float*)d_in[10];
    const float* W1     = (const float*)d_in[11];
    const float* b1     = (const float*)d_in[12];
    const float* ln_w_g = (const float*)d_in[13];
    const float* ln_w_b = (const float*)d_in[14];
    const float* W2     = (const float*)d_in[15];
    const float* b2     = (const float*)d_in[16];

    char* ws = (char*)d_ws;
    float*  q    = (float*)(ws);                         // 512*256 f32 = 512 KB
    float*  kmat = (float*)(ws + 524288);                // 512 KB
    float*  vmat = (float*)(ws + 1048576);               // 512 KB
    float*  dist = (float*)(ws + 1572864);               // 2 KB
    __bf16* w1f  = (__bf16*)(ws + 1574912);              // 128 KB (frag-swizzled)
    __bf16* w2f  = (__bf16*)(ws + 1574912 + 131072);     // 128 KB

    qkv_kernel<<<N_DIM, 256, 0, stream>>>(
        feat, Wq, bq, ln_q_g, ln_q_b, Wk, bk, ln_k_g, ln_k_b, Wv, bv,
        q, kmat, vmat, dist);

    wswz_kernel<<<512, 256, 0, stream>>>(W1, W2, w1f, w2f);

    cra_attn_kernel<<<N_DIM, 256, 0, stream>>>(
        q, kmat, vmat, dist, w1f, w2f, b1, ln_w_g, ln_w_b, b2, (float*)d_out);
}
